// PixelTwoStageOrdinalHead_39067022524474
// MI455X (gfx1250) — matrix operand order repaired and checked by reading, not compiled, not run on hardware
//
#include <hip/hip_runtime.h>

// ---------------- CDNA5 WMMA types ----------------
typedef __bf16 bf16_t;
typedef __attribute__((ext_vector_type(16))) bf16_t v16bf;
typedef __attribute__((ext_vector_type(8)))  float  v8f;

union Frag { v16bf v; uint4 q[2]; };

// ---------------- problem constants ----------------
#define BATCH 8
#define CIN   256
#define HH    128
#define WW    128
#define HW    16384
#define NOUT  512            // 256 damage + 256 severity channels fused
#define KTOT  2304           // 9 * 256
#define NKC   72             // KTOT / 32

// ---------------- workspace layout (bytes) ----------------
static constexpr size_t OFF_XT = 0;                      // bf16 x  NHWC [B][H][W][CIN]  64 MB
static constexpr size_t OFF_WB = 67108864;               // bf16 w  [NOUT][KTOT]         2.36 MB
static constexpr size_t OFF_H  = 69468160;               // bf16 h  [NOUT][B][H][W]      128 MB
static constexpr size_t OFF_ST = 203685888;              // f32 GN sums [B][16][2]
static constexpr size_t OFF_MR = OFF_ST + 1024;          // f32 mean/rstd [B][16][2]
static constexpr size_t OFF_MA = OFF_ST + 2048;          // f32 masked accum [B][5]
static constexpr size_t OFF_HC = OFF_ST + 4096;          // f32 hist cnt [B][4][256]
static constexpr size_t OFF_HS = OFF_ST + 36864;         // f32 hist sum [B][4][256]
#define NZERO 17408                                      // floats zeroed from OFF_ST

// ---------------- output offsets (floats, concat in return order) ----------------
#define OUT_CORN  131072
#define OUT_PIX   393216
#define OUT_PRED  917504
#define OUT_MEAN  1048576
#define OUT_TOPK  1048608
#define OUT_AGG   1048640
#define OUT_LBL   1048672

// ---------------- helpers ----------------
__device__ __forceinline__ unsigned short f2bf(float f) {
  union { float f; unsigned u; } v; v.f = f;
  return (unsigned short)((v.u + 0x7FFFu + ((v.u >> 16) & 1u)) >> 16);  // RNE
}
__device__ __forceinline__ float bf2f(unsigned short h) {
  union { unsigned u; float f; } v; v.u = ((unsigned)h) << 16; return v.f;
}
__device__ __forceinline__ float sigmoidf_(float x) { return 1.f / (1.f + expf(-x)); }
__device__ __forceinline__ float gelu_(float x) {
  return 0.5f * x * (1.f + erff(x * 0.7071067811865475f));
}

// ---------------- small utility kernels ----------------
__global__ void k_zero(float* __restrict__ p, int n) {
  int i = blockIdx.x * blockDim.x + threadIdx.x;
  if (i < n) p[i] = 0.f;
}

// NCHW f32 -> NHWC bf16, LDS-tiled so both phases are coalesced.
__global__ __launch_bounds__(256) void k_cvt_t(const float* __restrict__ x,
                                               unsigned short* __restrict__ xt) {
  __shared__ float tile[64][65];
  const int by = blockIdx.x;
  const int b  = by >> 7, y = by & 127;
  const int c0 = (blockIdx.y >> 1) * 64;
  const int x0 = (blockIdx.y & 1) * 64;
  const int t  = threadIdx.x;
  #pragma unroll
  for (int i = 0; i < 16; ++i) {
    const int idx = t + i * 256;
    const int cl = idx >> 6, xl = idx & 63;
    tile[cl][xl] = x[(((size_t)b * CIN + c0 + cl) * HH + y) * WW + x0 + xl];
  }
  __syncthreads();
  #pragma unroll
  for (int i = 0; i < 16; ++i) {
    const int idx = t + i * 256;
    const int xl = idx >> 6, cl = idx & 63;
    xt[(((size_t)b * HH + y) * WW + x0 + xl) * CIN + c0 + cl] = f2bf(tile[cl][xl]);
  }
}

// pack both conv3x3 weights into bf16 [NOUT][KTOT], K ordered (ky,kx,c)
__global__ void k_packw(const float* __restrict__ dw1, const float* __restrict__ sw1,
                        unsigned short* __restrict__ wb, int n) {
  int i = blockIdx.x * blockDim.x + threadIdx.x;
  if (i >= n) return;
  const int nn   = i / KTOT;
  const int k    = i - nn * KTOT;
  const int kpos = k >> 8;          // kernel tap 0..8
  const int c    = k & 255;
  const int ky   = kpos / 3;
  const int kx   = kpos - ky * 3;
  float v;
  if (nn < 256) v = dw1[(((size_t)nn * CIN + c) * 3 + ky) * 3 + kx];
  else          v = sw1[(((size_t)(nn - 256) * CIN + c) * 3 + ky) * 3 + kx];
  wb[i] = f2bf(v);
}

// ---------------- fused dual-head conv3x3 as implicit GEMM via WMMA ----------------
// grid: (B*H, NOUT/64); block: 256 threads = 8 waves
// block tile: M=128 (one image row) x N=64 (channels); K: 72 chunks of 32,
// double-buffered LDS with explicit gload -> ds_load frags -> 4x WMMA -> ds_store pipeline
__global__ __launch_bounds__(256) void k_conv(
    const unsigned short* __restrict__ xt,   // bf16 NHWC [B][H][W][CIN]
    const unsigned short* __restrict__ wb,   // bf16 [NOUT][KTOT]
    unsigned short* __restrict__ hout,       // bf16 [NOUT][B][H][W]
    float* __restrict__ stats)               // f32  [B][16][2] (sum, sumsq)
{
  __shared__ __align__(16) unsigned short sA[2][128][40];  // 80B rows, 16B padded
  __shared__ __align__(16) unsigned short sB[2][64][40];

  const int mb    = blockIdx.x;
  const int b     = mb >> 7;
  const int y     = mb & 127;
  const int ntile = blockIdx.y;
  const int t     = threadIdx.x;
  const int wave  = t >> 5;
  const int lane  = t & 31;

  v8f acc[4] = {};  // 4 x (16x16 f32) accumulators per wave

  // A-stage: thread -> (x row 0..127, 16-channel half); 2 x b128 per thread
  const int arow  = t >> 1;
  const int ahalf = t & 1;
  // B-stage: thread -> (n row, 16B part)
  const int nlB = t >> 2;
  const int ptB = t & 3;
  // fragment lane geometry (ISA 7.12.2, 16-bit A 16x32 / B 32x16 layouts)
  const int mrow = wave * 16 + (lane & 15);
  const int kbA  = (lane < 16) ? 0 : 8;     // A: k-pairs 0..7/16..23 vs 8..15/24..31
  const int ncol = lane & 15;
  const int kbB  = (lane < 16) ? 0 : 8;     // B: k 0..7/16..23 vs 8..15/24..31

  uint4 q0, q1, wv;  // staged registers (global -> regs -> LDS)

  auto gload = [&](int kc) {
    const int kk  = kc * 32;
    const int tap = kc >> 3;                // kernel tap 0..8
    const int c0  = (kc & 7) * 32;
    const int ky  = tap / 3;
    const int kx  = tap - ky * 3;
    const int gy  = y + ky - 1;
    const int gx  = arow + kx - 1;
    const bool inb = ((unsigned)gy < (unsigned)HH) && ((unsigned)gx < (unsigned)WW);
    q0 = (uint4){0, 0, 0, 0};
    q1 = (uint4){0, 0, 0, 0};
    if (inb) {
      const uint4* p = (const uint4*)(xt
          + (((size_t)b * HH + gy) * WW + gx) * CIN + c0 + ahalf * 16);
      q0 = p[0];
      q1 = p[1];
    }
    wv = *(const uint4*)(wb + (size_t)(ntile * 64 + nlB) * KTOT + kk + ptB * 8);
  };
  auto sstore = [&](int buf) {
    *(uint4*)(&sA[buf][arow][ahalf * 16])     = q0;
    *(uint4*)(&sA[buf][arow][ahalf * 16 + 8]) = q1;
    *(uint4*)(&sB[buf][nlB][ptB * 8])         = wv;
  };

  gload(0);
  sstore(0);
  __syncthreads();

  for (int kc = 0; kc < NKC; ++kc) {
    const int cur = kc & 1;
    if (kc + 1 < NKC) gload(kc + 1);        // global b128 loads in flight during math

    Frag a;
    a.q[0] = *(const uint4*)(&sA[cur][mrow][kbA]);
    a.q[1] = *(const uint4*)(&sA[cur][mrow][kbA + 16]);
    Frag bb[4];
    #pragma unroll
    for (int g = 0; g < 4; ++g) {
      bb[g].q[0] = *(const uint4*)(&sB[cur][g * 16 + ncol][kbB]);
      bb[g].q[1] = *(const uint4*)(&sB[cur][g * 16 + ncol][kbB + 16]);
    }
    #pragma unroll
    for (int g = 0; g < 4; ++g) {
      acc[g] = __builtin_amdgcn_wmma_f32_16x16x32_bf16(
          false, a.v, false, bb[g].v, (short)0, acc[g], false, false);
    }

    if (kc + 1 < NKC) sstore(cur ^ 1);      // LDS stores after the WMMA chain
    __syncthreads();
  }

  // epilogue: bf16 store of h + GroupNorm partial sums (group = n/32)
  const int x0 = wave * 16 + ((lane < 16) ? 0 : 8);
  float sum[2] = {0.f, 0.f}, ssq[2] = {0.f, 0.f};
  #pragma unroll
  for (int g = 0; g < 4; ++g) {
    const int n = ntile * 64 + g * 16 + ncol;
    unsigned pk[4];
    #pragma unroll
    for (int j = 0; j < 4; ++j) {
      float v0 = acc[g][2 * j];
      float v1 = acc[g][2 * j + 1];
      sum[g >> 1] += v0 + v1;
      ssq[g >> 1] += v0 * v0 + v1 * v1;
      pk[j] = (unsigned)f2bf(v0) | ((unsigned)f2bf(v1) << 16);
    }
    uint4 o; o.x = pk[0]; o.y = pk[1]; o.z = pk[2]; o.w = pk[3];
    *(uint4*)(hout + (((size_t)n * BATCH + b) * HH + y) * WW + x0) = o;
  }
  #pragma unroll
  for (int off = 16; off > 0; off >>= 1) {
    sum[0] += __shfl_xor(sum[0], off, 32);
    ssq[0] += __shfl_xor(ssq[0], off, 32);
    sum[1] += __shfl_xor(sum[1], off, 32);
    ssq[1] += __shfl_xor(ssq[1], off, 32);
  }
  if (lane == 0) {
    const int gA = ntile * 2;
    atomicAdd(&stats[(b * 16 + gA) * 2 + 0], sum[0]);
    atomicAdd(&stats[(b * 16 + gA) * 2 + 1], ssq[0]);
    atomicAdd(&stats[(b * 16 + gA + 1) * 2 + 0], sum[1]);
    atomicAdd(&stats[(b * 16 + gA + 1) * 2 + 1], ssq[1]);
  }
}

// ---------------- finalize GroupNorm stats ----------------
__global__ void k_gnstats(const float* __restrict__ sums, float* __restrict__ mr) {
  int i = blockIdx.x * blockDim.x + threadIdx.x;
  if (i >= BATCH * 16) return;
  const float N = 32.f * (float)HW;
  float s = sums[2 * i], q = sums[2 * i + 1];
  float m = s / N;
  float var = q / N - m * m;
  mr[2 * i]     = m;
  mr[2 * i + 1] = rsqrtf(var + 1e-5f);
}

// ---------------- GN + GELU + 1x1 convs + CORN decode + masked accum ----------------
__global__ __launch_bounds__(128) void k_pointwise(
    const unsigned short* __restrict__ h, const float* __restrict__ mr,
    const float* __restrict__ dgs, const float* __restrict__ dgb,
    const float* __restrict__ dw2, const float* __restrict__ db2,
    const float* __restrict__ sgs, const float* __restrict__ sgb,
    const float* __restrict__ sw2, const float* __restrict__ sb2,
    const float* __restrict__ mask, float* __restrict__ out,
    float* __restrict__ macc, float* __restrict__ hcnt, float* __restrict__ hsum)
{
  const int mb = blockIdx.x;
  const int b = mb >> 7, y = mb & 127;
  const int x = threadIdx.x;
  const size_t yx    = (size_t)y * WW + x;
  const size_t pixix = (size_t)b * HW + yx;

  float dacc = 0.f, s0a = 0.f, s1a = 0.f;
  for (int c = 0; c < 256; ++c) {
    const int gd = b * 16 + (c >> 5);
    const int gs = gd + 8;
    float hd = bf2f(h[((size_t)c * BATCH + b) * HW + yx]);
    float hs = bf2f(h[((size_t)(c + 256) * BATCH + b) * HW + yx]);
    float vd = (hd - mr[2 * gd]) * mr[2 * gd + 1] * dgs[c] + dgb[c];
    vd = gelu_(vd);
    dacc += dw2[c] * vd;
    float vs = (hs - mr[2 * gs]) * mr[2 * gs + 1] * sgs[c] + sgb[c];
    vs = gelu_(vs);
    s0a += sw2[c] * vs;
    s1a += sw2[256 + c] * vs;
  }
  const float dl = dacc + db2[0];
  const float c0 = s0a + sb2[0];
  const float c1 = s1a + sb2[1];

  // CORN dense decode
  const float t0 = sigmoidf_(c0);
  const float t1 = t0 * sigmoidf_(c1);
  float sev[3] = {1.f - t0, t0 - t1, t1};
  float ssum = 0.f;
  for (int j = 0; j < 3; ++j) { sev[j] = fmaxf(sev[j], 1e-8f); ssum += sev[j]; }
  ssum = fmaxf(ssum, 1e-8f);
  for (int j = 0; j < 3; ++j) sev[j] /= ssum;

  const float p = sigmoidf_(dl);
  float pix[4] = {1.f - p, p * sev[0], p * sev[1], p * sev[2]};
  float psum = 0.f;
  for (int j = 0; j < 4; ++j) { pix[j] = fmaxf(pix[j], 1e-8f); psum += pix[j]; }
  psum = fmaxf(psum, 1e-8f);
  for (int j = 0; j < 4; ++j) pix[j] /= psum;

  int si = 0; float sbest = sev[0];
  if (sev[1] > sbest) { si = 1; sbest = sev[1]; }
  if (sev[2] > sbest) { si = 2; }
  const int pred = (p >= 0.5f) ? (si + 1) : 0;

  out[pixix] = dl;
  out[OUT_CORN + ((size_t)b * 2 + 0) * HW + yx] = c0;
  out[OUT_CORN + ((size_t)b * 2 + 1) * HW + yx] = c1;
  for (int j = 0; j < 4; ++j)
    out[OUT_PIX + ((size_t)b * 4 + j) * HW + yx] = pix[j];
  out[OUT_PRED + pixix] = (float)pred;

  if (mask[pixix] > 0.5f) {
    atomicAdd(&macc[b * 5 + 4], 1.f);
    for (int j = 0; j < 4; ++j) {
      atomicAdd(&macc[b * 5 + j], pix[j]);
      int bin = (int)(pix[j] * 256.f);
      bin = bin < 0 ? 0 : (bin > 255 ? 255 : bin);
      atomicAdd(&hcnt[((size_t)b * 4 + j) * 256 + bin], 1.f);
      atomicAdd(&hsum[((size_t)b * 4 + j) * 256 + bin], pix[j]);
    }
  }
}

// ---------------- per-batch aggregation (mean / topk / agg) ----------------
__global__ void k_finalize(const float* __restrict__ macc,
                           const float* __restrict__ hcnt,
                           const float* __restrict__ hsum,
                           float* __restrict__ out)
{
  const int b = blockIdx.x;
  if (threadIdx.x != 0) return;
  const float cnt   = macc[b * 5 + 4];
  const bool  empty = cnt < 0.5f;
  const float denom = fmaxf(cnt, 1.f);
  float kf = rintf(cnt * 0.2f);
  kf = fminf(fmaxf(kf, 1.f), denom);

  float mean[4], topk[4];
  for (int c = 0; c < 4; ++c)
    mean[c] = empty ? 0.25f : macc[b * 5 + c] / denom;
  float ms = 0.f;
  for (int c = 0; c < 4; ++c) { mean[c] = fmaxf(mean[c], 1e-8f); ms += mean[c]; }
  ms = fmaxf(ms, 1e-8f);
  for (int c = 0; c < 4; ++c) mean[c] /= ms;

  for (int c = 0; c < 4; ++c) {
    if (empty) { topk[c] = 0.25f; continue; }
    float cacc = 0.f, sacc = 0.f;
    for (int bin = 255; bin >= 0; --bin) {
      const float hc = hcnt[((size_t)b * 4 + c) * 256 + bin];
      const float hs = hsum[((size_t)b * 4 + c) * 256 + bin];
      if (cacc + hc >= kf) {
        const float need = kf - cacc;
        sacc += hs * (need / fmaxf(hc, 1e-12f));
        break;
      }
      cacc += hc; sacc += hs;
    }
    topk[c] = sacc / kf;
  }
  float ts = 0.f;
  for (int c = 0; c < 4; ++c) { topk[c] = fmaxf(topk[c], 1e-8f); ts += topk[c]; }
  ts = fmaxf(ts, 1e-8f);
  for (int c = 0; c < 4; ++c) topk[c] /= ts;

  float agg[4], as = 0.f;
  for (int c = 0; c < 4; ++c) {
    agg[c] = fmaxf(0.7f * mean[c] + 0.3f * topk[c], 1e-8f);
    as += agg[c];
  }
  as = fmaxf(as, 1e-8f);
  for (int c = 0; c < 4; ++c) agg[c] /= as;
  int ai = 0; float ab = agg[0];
  for (int c = 1; c < 4; ++c) if (agg[c] > ab) { ab = agg[c]; ai = c; }

  for (int c = 0; c < 4; ++c) {
    out[OUT_MEAN + b * 4 + c] = mean[c];
    out[OUT_TOPK + b * 4 + c] = topk[c];
    out[OUT_AGG  + b * 4 + c] = agg[c];
  }
  out[OUT_LBL + b] = (float)ai;
}

// ---------------- launch ----------------
extern "C" void kernel_launch(void* const* d_in, const int* in_sizes, int n_in,
                              void* d_out, int out_size, void* d_ws, size_t ws_size,
                              hipStream_t stream) {
  (void)in_sizes; (void)n_in; (void)out_size; (void)ws_size;
  const float* x   = (const float*)d_in[0];
  const float* msk = (const float*)d_in[1];
  const float* dw1 = (const float*)d_in[2];
  const float* dgs = (const float*)d_in[3];
  const float* dgb = (const float*)d_in[4];
  const float* dw2 = (const float*)d_in[5];
  const float* db2 = (const float*)d_in[6];
  const float* sw1 = (const float*)d_in[7];
  const float* sgs = (const float*)d_in[8];
  const float* sgb = (const float*)d_in[9];
  const float* sw2 = (const float*)d_in[10];
  const float* sb2 = (const float*)d_in[11];
  float* out = (float*)d_out;
  char*  ws  = (char*)d_ws;

  unsigned short* xt = (unsigned short*)(ws + OFF_XT);
  unsigned short* wb = (unsigned short*)(ws + OFF_WB);
  unsigned short* h  = (unsigned short*)(ws + OFF_H);
  float* stats = (float*)(ws + OFF_ST);
  float* mr    = (float*)(ws + OFF_MR);
  float* macc  = (float*)(ws + OFF_MA);
  float* hcnt  = (float*)(ws + OFF_HC);
  float* hsum  = (float*)(ws + OFF_HS);

  k_zero<<<(NZERO + 255) / 256, 256, 0, stream>>>(stats, NZERO);
  k_cvt_t<<<dim3(BATCH * HH, 8), 256, 0, stream>>>(x, xt);
  {
    const int n = NOUT * KTOT;
    k_packw<<<(n + 255) / 256, 256, 0, stream>>>(dw1, sw1, wb, n);
  }
  k_conv<<<dim3(BATCH * HH, NOUT / 64), 256, 0, stream>>>(xt, wb, h, stats);
  k_gnstats<<<1, 128, 0, stream>>>(stats, mr);
  k_pointwise<<<BATCH * HH, 128, 0, stream>>>(h, mr, dgs, dgb, dw2, db2,
                                              sgs, sgb, sw2, sb2, msk, out,
                                              macc, hcnt, hsum);
  k_finalize<<<BATCH, 32, 0, stream>>>(macc, hcnt, hsum, out);
}
